// Int8Linear_29669634081272
// MI455X (gfx1250) — matrix operand order repaired and checked by reading, not compiled, not run on hardware
//
#include <hip/hip_runtime.h>

// Int8-weight linear: out[M,N] = x[M,K] @ Wq[N,K]^T * scale[N]
// M = 8192 (B*S), K = 4096, N = 11008.  Memory-bound (~675 MB @ 23.3 TB/s).
// f16 WMMA (V_WMMA_F32_16X16X32_F16), f32 accumulate, per-channel scale epilogue.

#define M_DIM 8192
#define N_DIM 11008
#define K_DIM 4096
#define BM 128
#define BN 128
#define BK 64
#define LDA 72   // 64 + 8 halfword pad -> 144B row stride = 36 banks (conflict-free)
#define LDB 72

typedef __attribute__((ext_vector_type(16))) _Float16 v16h;
typedef __attribute__((ext_vector_type(8)))  _Float16 v8h;
typedef __attribute__((ext_vector_type(4)))  _Float16 h4;
typedef __attribute__((ext_vector_type(8)))  float    v8f;
typedef __attribute__((ext_vector_type(4)))  float    f32x4;
typedef __attribute__((ext_vector_type(4)))  int      i32x4;

union Frag16 { v16h f; v8h h[2]; };

__global__ __launch_bounds__(256)
void int8_linear_wmma(const float* __restrict__ x,
                      const int*   __restrict__ wq,
                      const float* __restrict__ scale,
                      float*       __restrict__ out)
{
    __shared__ __align__(16) _Float16 As[BM * LDA];
    __shared__ __align__(16) _Float16 Bs[BN * LDB];

    const int tid    = threadIdx.x;
    const int lane   = tid & 31;
    const int wave   = tid >> 5;
    const int wm     = wave >> 1;      // 0..3  (M direction)
    const int wn     = wave & 1;       // 0..1  (N direction)
    const int laneN  = lane & 15;
    const int laneHi = lane >> 4;      // 0/1

    // Grid: x = M tiles (fastest) so one 2MB B-stripe is L2-hot per generation
    // while all of x (134MB) stays resident in the 192MB L2.
    const int mBlock = blockIdx.x * BM;
    const int nBlock = blockIdx.y * BN;

    const float* xg = x  + (size_t)mBlock * K_DIM;
    const int*   wg = wq + (size_t)nBlock * K_DIM;

    v8f acc[2][4];
#pragma unroll
    for (int fm = 0; fm < 2; ++fm)
#pragma unroll
        for (int fn = 0; fn < 4; ++fn)
            acc[fm][fn] = (v8f)(0.0f);

    for (int k0 = 0; k0 < K_DIM; k0 += BK) {
        // ---- Global -> registers (convert fp32/int32 -> f16) -> LDS ----
        // 128 rows x 64 k-elements per tile; 256 threads x 8 iters x 4 elems.
#pragma unroll
        for (int j = 0; j < 8; ++j) {
            const int    idx  = tid + j * 256;
            const int    row  = idx >> 4;           // 0..127
            const int    kq   = (idx & 15) << 2;    // 0,4,...,60
            const size_t goff = (size_t)row * K_DIM + k0 + kq;

            f32x4 xv = *(const f32x4*)(xg + goff);
            i32x4 wv = __builtin_nontemporal_load((const i32x4*)(wg + goff));

            if (k0 + BK < K_DIM) {                  // hint next K tile
                __builtin_prefetch(xg + goff + BK, 0, 3);
                __builtin_prefetch(wg + goff + BK, 0, 0);
            }

            h4 xh, wh;
            xh[0] = (_Float16)xv[0]; xh[1] = (_Float16)xv[1];
            xh[2] = (_Float16)xv[2]; xh[3] = (_Float16)xv[3];
            wh[0] = (_Float16)wv[0]; wh[1] = (_Float16)wv[1];
            wh[2] = (_Float16)wv[2]; wh[3] = (_Float16)wv[3];

            *(h4*)(&As[row * LDA + kq]) = xh;
            *(h4*)(&Bs[row * LDB + kq]) = wh;
        }
        __syncthreads();

        // ---- Two WMMA k-steps (K=32 each) over the LDS tile ----
#pragma unroll
        for (int ks = 0; ks < 2; ++ks) {
            const int kk = ks * 32;
            Frag16 a[2], b[4];

            // A 16x32 f16 layout: lane holds row m=lane%16, two contiguous
            // 8-element K runs at K = 8*(lane/16) and K = 16 + 8*(lane/16).
            const int ca = laneHi * 8;
#pragma unroll
            for (int fm = 0; fm < 2; ++fm) {
                const _Float16* ap =
                    &As[(wm * 32 + fm * 16 + laneN) * LDA + kk + ca];
                a[fm].h[0] = *(const v8h*)(ap);
                a[fm].h[1] = *(const v8h*)(ap + 16);
            }

            // B 32x16 f16 layout: lane holds column n=lane%16, two contiguous
            // 8-element K runs at K = 8*(lane/16) and K = 16 + 8*(lane/16).
            const int cb = laneHi * 8;
#pragma unroll
            for (int fn = 0; fn < 4; ++fn) {
                const _Float16* bp =
                    &Bs[(wn * 64 + fn * 16 + laneN) * LDB + kk + cb];
                b[fn].h[0] = *(const v8h*)(bp);
                b[fn].h[1] = *(const v8h*)(bp + 16);
            }

#pragma unroll
            for (int fm = 0; fm < 2; ++fm)
#pragma unroll
                for (int fn = 0; fn < 4; ++fn)
                    acc[fm][fn] = __builtin_amdgcn_wmma_f32_16x16x32_f16(
                        false, a[fm].f, false, b[fn].f,
                        (short)0, acc[fm][fn], false, false);
        }
        __syncthreads();
    }

    // ---- Epilogue: per-output-channel scale, non-temporal streaming store ----
    // C/D layout: VGPR i -> (M = i + 8*laneHi, N = lane%16) within the 16x16 frag.
#pragma unroll
    for (int fn = 0; fn < 4; ++fn) {
        const int   n = nBlock + wn * 64 + fn * 16 + laneN;
        const float s = scale[n];
#pragma unroll
        for (int fm = 0; fm < 2; ++fm) {
            const int m0 = mBlock + wm * 32 + fm * 16 + laneHi * 8;
            float* op = out + (size_t)m0 * N_DIM + n;
            v8f c = acc[fm][fn];
#pragma unroll
            for (int i = 0; i < 8; ++i)
                __builtin_nontemporal_store(c[i] * s, op + (size_t)i * N_DIM);
        }
    }
}

extern "C" void kernel_launch(void* const* d_in, const int* in_sizes, int n_in,
                              void* d_out, int out_size, void* d_ws, size_t ws_size,
                              hipStream_t stream) {
    const float* x     = (const float*)d_in[0];   // [8192, 4096] fp32
    const int*   wq    = (const int*)  d_in[1];   // [11008, 4096] int32 (int8 values)
    const float* scale = (const float*)d_in[2];   // [11008] fp32
    float*       out   = (float*)d_out;           // [8192, 11008] fp32

    dim3 grid(M_DIM / BM, N_DIM / BN, 1);         // 64 x 86 blocks
    dim3 block(256, 1, 1);                        // 8 wave32 waves
    int8_linear_wmma<<<grid, block, 0, stream>>>(x, wq, scale, out);
}
